// MultiHeadDotProductAttention_24223615549863
// MI455X (gfx1250) — matrix operand order repaired and checked by reading, not compiled, not run on hardware
//
#include <hip/hip_runtime.h>
#include <hip/hip_bf16.h>

typedef __attribute__((ext_vector_type(16))) _Float16 v16h;
typedef __attribute__((ext_vector_type(8)))  _Float16 v8h;
typedef __attribute__((ext_vector_type(8)))  float    v8f;
typedef __attribute__((ext_vector_type(4)))  unsigned v4u;
typedef __attribute__((ext_vector_type(8)))  unsigned v8u;

#define B_   16
#define L_   1025
#define LP   1088          // padded seq len (mult of 64)
#define F_   768
#define H_   12
#define D_   64
#define NROW (B_ * L_)     // 16400 = 1025 * 16 row-tiles exactly
#define C3   (3 * F_)      // 2304
#define EPSF 1e-8f

// ---------------- WMMA helpers (gfx1250 wave32) ----------------

__device__ __forceinline__ v8f wmma16(v16h a, v16h b, v8f c) {
  // v_wmma_f32_16x16x32_f16
  return __builtin_amdgcn_wmma_f32_16x16x32_f16(false, a, false, b, (short)0, c,
                                                false, false);
}

// A-operand fragment (16x32 f16). `row` points at this lane's matrix row
// (m = lane&15), d0 = K-chunk base, hi = lane>>4.
// Element e of v16h maps to k = d0 + hi*16 + e.
__device__ __forceinline__ v16h load_a(const _Float16* row, int d0, int hi) {
  v8h lo = *(const v8h*)(row + d0 + hi * 16);
  v8h hm = *(const v8h*)(row + d0 + hi * 16 + 8);
  v16h r;
#pragma unroll
  for (int i = 0; i < 8; ++i) { r[i] = lo[i]; r[i + 8] = hm[i]; }
  return r;
}
// B-operand fragment (32x16 f16): lane n = lane&15, element e -> k = hi*16+e.
__device__ __forceinline__ v16h load_b(const _Float16* kvec, int hi) {
  return *(const v16h*)(kvec + hi * 16);
}

// CDNA5 async DMA: global -> LDS, 16B per lane, tracked by ASYNCcnt.
__device__ __forceinline__ void async_b128(void* lds_dst, const void* gsrc) {
  uint32_t l = (uint32_t)(uintptr_t)lds_dst;  // flat LDS addr[31:0] = offset
  asm volatile("global_load_async_to_lds_b128 %0, %1, off"
               :: "v"(l), "v"(gsrc)
               : "memory");
}
__device__ __forceinline__ void wait_async0() {
  asm volatile("s_wait_asynccnt 0" ::: "memory");
}

// CDNA5 Tensor Data Mover: 2-D tile load, D# groups 0/1 in SGPR tuples
// (VADDR2/3 = NULL -> tensors up to 2D). Tracked by TENSORcnt.
__device__ __forceinline__ void tdm_load_2d(unsigned lds_addr, uint64_t gaddr,
                                            v8u g1) {
  v4u g0;
  g0[0] = 1u;                                   // count=1 (valid user desc)
  g0[1] = lds_addr;                             // LDS byte address
  g0[2] = (unsigned)gaddr;                      // global_addr[31:0]
  g0[3] = ((unsigned)(gaddr >> 32) & 0x01FFFFFFu) | 0x80000000u; // [56:32]|type=2
  asm volatile("tensor_load_to_lds %0, %1" :: "s"(g0), "s"(g1) : "memory");
}

// ---------------- prep kernels ----------------

__global__ void cvt_x_kernel(const float* __restrict__ x,
                             _Float16* __restrict__ xh, int n) {
  int i = blockIdx.x * blockDim.x + threadIdx.x;
  int s = gridDim.x * blockDim.x;
  for (; i < n; i += s) xh[i] = (_Float16)x[i];
}

// wt[col][f] = w_seg[f][col%768] ; col = seg*768 + h*64 + d  (K-dim contiguous)
__global__ void prep_wqkv_kernel(const float* __restrict__ wq,
                                 const float* __restrict__ wk,
                                 const float* __restrict__ wv,
                                 _Float16* __restrict__ wt) {
  int i = blockIdx.x * blockDim.x + threadIdx.x;
  int n = C3 * F_;
  int s = gridDim.x * blockDim.x;
  for (; i < n; i += s) {
    int col = i / F_, f = i % F_;
    int seg = col / F_, inner = col % F_;
    const float* src = seg == 0 ? wq : (seg == 1 ? wk : wv);
    wt[i] = (_Float16)src[f * F_ + inner];
  }
}

__global__ void prep_bias_kernel(const float* __restrict__ bq,
                                 const float* __restrict__ bk,
                                 const float* __restrict__ bv,
                                 float* __restrict__ bqkv) {
  int i = blockIdx.x * blockDim.x + threadIdx.x;
  if (i >= C3) return;
  int seg = i / F_, inner = i % F_;
  const float* src = seg == 0 ? bq : (seg == 1 ? bk : bv);
  bqkv[i] = src[inner];
}

// woT[fout][hd] = wo[hd][fout]  (wo given as [H,D,F] = [768][768])
__global__ void prep_wo_kernel(const float* __restrict__ wo,
                               _Float16* __restrict__ woT) {
  int i = blockIdx.x * blockDim.x + threadIdx.x;
  int n = F_ * F_;
  int s = gridDim.x * blockDim.x;
  for (; i < n; i += s) {
    int fout = i / F_, hd = i % F_;
    woT[i] = (_Float16)wo[hd * F_ + fout];
  }
}

// ---------------- fused QKV projection + activation ----------------
__global__ __launch_bounds__(128) void qkv_gemm_kernel(
    const _Float16* __restrict__ xh, const _Float16* __restrict__ wt,
    const float* __restrict__ bias, _Float16* __restrict__ q16,
    _Float16* __restrict__ k16, _Float16* __restrict__ vT) {
  const int lane = threadIdx.x & 31;
  const int wave = threadIdx.x >> 5;
  const int m = lane & 15, hi = lane >> 4;
  const int rowblk = blockIdx.x;                 // 0..1024
  const int colblk = blockIdx.y * 4 + wave;      // 0..143
  const int col = colblk * 16 + m;
  const _Float16* arow = xh + (size_t)(rowblk * 16 + m) * F_;
  const _Float16* brow = wt + (size_t)col * F_;
  v8f acc = {};
#pragma unroll 4
  for (int f = 0; f < F_; f += 32) {
    v16h a = load_a(arow, f, hi);
    v16h b = load_b(brow + f, hi);
    acc = wmma16(a, b, acc);
  }
  const float bb = bias[col];
  const int seg = col / F_;
  const int hh = (col % F_) / D_;
  const int dd = col % D_;
#pragma unroll
  for (int r = 0; r < 8; ++r) {
    int row = rowblk * 16 + r + hi * 8;          // < 16400 always
    int bi = row / L_, li = row % L_;
    float v = acc[r] + bb;
    if (seg == 0) {                               // q: relu(q/sqrt(64)) + eps
      v = fmaxf(v * 0.125f, 0.f) + EPSF;
      q16[((size_t)(bi * H_ + hh) * LP + li) * D_ + dd] = (_Float16)v;
    } else if (seg == 1) {                        // k: relu(k) + eps
      v = fmaxf(v, 0.f) + EPSF;
      k16[((size_t)(bi * H_ + hh) * LP + li) * D_ + dd] = (_Float16)v;
    } else {                                      // v, stored transposed
      vT[((size_t)(bi * H_ + hh) * D_ + dd) * LP + li] = (_Float16)v;
    }
  }
}

// ---------------- streaming masked attention ----------------
// grid = (17 q-blocks of 64, H, B); 4 waves, each owns a 16-query tile.
// K/V tiles (32 keys) staged in LDS by the Tensor Data Mover, double-buffered;
// wave 0 issues TDM descriptors, TENSORcnt + barrier publishes to all waves.
__global__ __launch_bounds__(128) void attn_kernel(
    const _Float16* __restrict__ q16, const _Float16* __restrict__ k16,
    const _Float16* __restrict__ vT, const float* __restrict__ toeplitz,
    _Float16* __restrict__ aout) {
  __shared__ __align__(64) float tm_s[4096];          // raw toeplitz_params[h]
  __shared__ __align__(64) _Float16 kt_s[2][32 * 64]; // [key][d]
  __shared__ __align__(64) _Float16 vt_s[2][64 * 32]; // [d][key]
  __shared__ __align__(64) _Float16 p_s[4][16 * 32];

  const int b = blockIdx.z, h = blockIdx.y, qb = blockIdx.x;
  const int wave = threadIdx.x >> 5, lane = threadIdx.x & 31;
  const int m = lane & 15, hi = lane >> 4;
  const int tid = threadIdx.x;

  // stage toeplitz row (16KB) via per-lane async DMA; |.| folded into fma later
  const float* tp = toeplitz + (size_t)h * 4096;
#pragma unroll
  for (int j = 0; j < 8; ++j) {
    int c = tid + 128 * j;
    async_b128(&tm_s[c * 4], tp + c * 4);
  }

  const int q0 = qb * 64 + wave * 16;                 // may exceed L_, padded
  const _Float16* qrow = q16 + ((size_t)(b * H_ + h) * LP + q0 + m) * D_;
  const _Float16* kbase = k16 + (size_t)(b * H_ + h) * LP * D_;
  const _Float16* vbase = vT + (size_t)(b * H_ + h) * D_ * LP;
  _Float16* pw = &p_s[wave][0];

  v16h aq0 = load_a(qrow, 0, hi);
  v16h aq1 = load_a(qrow, 32, hi);

  // per-row toeplitz constants: idx = Cq[r] - (2*pk - (pk&31))
  int Cq[8];
  bool qz[8];
#pragma unroll
  for (int r = 0; r < 8; ++r) {
    int qi = q0 + r + hi * 8;
    qz[r] = (qi == 0) || (qi >= L_);
    int pq = qi - 1;
    Cq[r] = 2 * pq - (pq & 31) + 2080;
  }

  // TDM issuing wave: must be a true scalar branch (TDM ignores EXEC)
  const bool issuer = (__builtin_amdgcn_readfirstlane(wave) == 0);

  // D# group1 descriptors (wave-uniform constants)
  // K tile: 32 rows x 64 elems, row stride 64, 2B elements
  const v8u g1k = {0x00010000u,        // data_size = 2B
                   64u << 16,          // tensor_dim0 = 64
                   (unsigned)LP << 16, // tensor_dim1 = LP
                   64u << 16,          // tile_dim0 = 64
                   32u,                // tile_dim1 = 32 (tile_dim2 = 0)
                   64u,                // tensor_dim0_stride = 64
                   0u, 0u};
  // V tile: 64 rows x 32 elems, row stride LP, 2B elements
  const v8u g1v = {0x00010000u,
                   (unsigned)LP << 16, // tensor_dim0 = LP
                   64u << 16,          // tensor_dim1 = 64
                   32u << 16,          // tile_dim0 = 32
                   64u,                // tile_dim1 = 64
                   (unsigned)LP,       // tensor_dim0_stride = LP
                   0u, 0u};

  auto stage = [&](int buf, int kb) {
    tdm_load_2d((unsigned)(uintptr_t)&kt_s[buf][0],
                (uint64_t)(uintptr_t)(kbase + (size_t)kb * D_), g1k);
    tdm_load_2d((unsigned)(uintptr_t)&vt_s[buf][0],
                (uint64_t)(uintptr_t)(vbase + kb), g1v);
  };

  v8f o0 = {}, o1 = {}, o2 = {}, o3 = {};
  float rsum[8] = {0.f, 0.f, 0.f, 0.f, 0.f, 0.f, 0.f, 0.f};

  if (issuer) stage(0, 0);
  wait_async0();                                      // tm_s resident
  if (issuer) __builtin_amdgcn_s_wait_tensorcnt(0);
  __syncthreads();

  for (int it = 0; it < 33; ++it) {                   // keys padded to 1056
    const int kb = it * 32;
    const int buf = it & 1;
    if (issuer && it + 1 < 33) stage(buf ^ 1, kb + 32); // overlap DMA/compute

    const _Float16* kt = &kt_s[buf][0];
    const _Float16* vt = &vt_s[buf][0];

    v8f s0 = {}, s1 = {};
    s0 = wmma16(aq0, load_b(kt + m * 64, hi), s0);
    s0 = wmma16(aq1, load_b(kt + m * 64 + 32, hi), s0);
    s1 = wmma16(aq0, load_b(kt + (16 + m) * 64, hi), s1);
    s1 = wmma16(aq1, load_b(kt + (16 + m) * 64 + 32, hi), s1);

    // mask + eps in C-layout: element r -> (mq = r + 8*hi, n = lane&15)
    const int ki0 = kb + m, ki1 = ki0 + 16;
    const bool kv0 = ki0 < L_, kv1 = ki1 < L_;
    const bool kz0 = (ki0 == 0);
    const int Tk0 = 2 * (ki0 - 1) - ((ki0 - 1) & 31);
    const int Tk1 = 2 * (ki1 - 1) - ((ki1 - 1) & 31);
#pragma unroll
    for (int r = 0; r < 8; ++r) {
      float w0 = (qz[r] || kz0) ? 1.f : tm_s[(Cq[r] - Tk0) & 4095];
      float w1 = qz[r] ? 1.f : tm_s[(Cq[r] - Tk1) & 4095];
      float p0 = kv0 ? fmaf(s0[r], fabsf(w0), EPSF) : 0.f;  // |w| = src mod
      float p1 = kv1 ? fmaf(s1[r], fabsf(w1), EPSF) : 0.f;
      rsum[r] += p0 + p1;
      int mq = r + hi * 8;
      pw[mq * 32 + m] = (_Float16)p0;
      pw[mq * 32 + 16 + m] = (_Float16)p1;
    }
    asm volatile("s_wait_dscnt 0" ::: "memory");      // same-wave LDS RAW

    v16h ap = load_a(pw + m * 32, 0, hi);             // P as 16x32 A fragment
    o0 = wmma16(ap, load_b(vt + (0 * 16 + m) * 32, hi), o0);
    o1 = wmma16(ap, load_b(vt + (1 * 16 + m) * 32, hi), o1);
    o2 = wmma16(ap, load_b(vt + (2 * 16 + m) * 32, hi), o2);
    o3 = wmma16(ap, load_b(vt + (3 * 16 + m) * 32, hi), o3);

    if (issuer) __builtin_amdgcn_s_wait_tensorcnt(0); // next tile resident
    __syncthreads();                                  // all waves done w/ buf
  }

  // reduce rowsum over the 16 n-lanes of this hi-group
#pragma unroll
  for (int r = 0; r < 8; ++r) {
    float s = rsum[r];
    s += __shfl_xor(s, 1, 32);
    s += __shfl_xor(s, 2, 32);
    s += __shfl_xor(s, 4, 32);
    s += __shfl_xor(s, 8, 32);
    rsum[r] = s;
  }
  // store normalized output: aout[B, L, H*D] f16 (rows match final GEMM)
#pragma unroll
  for (int dblk = 0; dblk < 4; ++dblk) {
    const v8f o = dblk == 0 ? o0 : dblk == 1 ? o1 : dblk == 2 ? o2 : o3;
#pragma unroll
    for (int r = 0; r < 8; ++r) {
      int qi = q0 + r + hi * 8;
      if (qi < L_)
        aout[(size_t)(b * L_ + qi) * F_ + h * D_ + dblk * 16 + m] =
            (_Float16)(o[r] / rsum[r]);
    }
  }
}

// ---------------- output projection ----------------
__global__ __launch_bounds__(128) void out_gemm_kernel(
    const _Float16* __restrict__ ah, const _Float16* __restrict__ woT,
    const float* __restrict__ bo, float* __restrict__ out) {
  const int lane = threadIdx.x & 31;
  const int wave = threadIdx.x >> 5;
  const int m = lane & 15, hi = lane >> 4;
  const int rowblk = blockIdx.x;
  const int colblk = blockIdx.y * 4 + wave;        // 0..47
  const int col = colblk * 16 + m;
  const _Float16* arow = ah + (size_t)(rowblk * 16 + m) * F_;
  const _Float16* brow = woT + (size_t)col * F_;
  v8f acc = {};
#pragma unroll 4
  for (int f = 0; f < F_; f += 32) {
    v16h a = load_a(arow, f, hi);
    v16h b = load_b(brow + f, hi);
    acc = wmma16(a, b, acc);
  }
  const float bb = bo[col];
#pragma unroll
  for (int r = 0; r < 8; ++r) {
    int row = rowblk * 16 + r + hi * 8;
    out[(size_t)row * F_ + col] = acc[r] + bb;
  }
}

// ---------------- launch ----------------
extern "C" void kernel_launch(void* const* d_in, const int* in_sizes, int n_in,
                              void* d_out, int out_size, void* d_ws,
                              size_t ws_size, hipStream_t stream) {
  const float* x  = (const float*)d_in[0];
  const float* wq = (const float*)d_in[1];
  const float* bq = (const float*)d_in[2];
  const float* wk = (const float*)d_in[3];
  const float* bk = (const float*)d_in[4];
  const float* wv = (const float*)d_in[5];
  const float* bv = (const float*)d_in[6];
  const float* wo = (const float*)d_in[7];
  const float* bo = (const float*)d_in[8];
  const float* tz = (const float*)d_in[9];
  float* out = (float*)d_out;

  size_t off = 0;
  auto carve = [&](size_t bytes) -> void* {
    void* p = (char*)d_ws + off;
    off += (bytes + 255) & ~(size_t)255;
    return p;
  };
  const size_t QKV_BYTES = (size_t)B_ * H_ * LP * D_ * sizeof(_Float16);
  _Float16* xh   = (_Float16*)carve((size_t)NROW * F_ * 2);
  _Float16* wt   = (_Float16*)carve((size_t)C3 * F_ * 2);
  float*    bqkv = (float*)carve((size_t)C3 * 4);
  _Float16* woT  = (_Float16*)carve((size_t)F_ * F_ * 2);
  _Float16* q16  = (_Float16*)carve(QKV_BYTES);
  _Float16* k16  = (_Float16*)carve(QKV_BYTES);
  _Float16* vTd  = (_Float16*)carve(QKV_BYTES);
  _Float16* aout = (_Float16*)carve((size_t)NROW * F_ * 2);

  // zero q/k/v (covers L padding rows); buffers are contiguous & 256B-sized
  hipMemsetAsync(q16, 0, 3 * QKV_BYTES, stream);

  cvt_x_kernel<<<4096, 256, 0, stream>>>(x, xh, NROW * F_);
  prep_wqkv_kernel<<<2048, 256, 0, stream>>>(wq, wk, wv, wt);
  prep_bias_kernel<<<(C3 + 255) / 256, 256, 0, stream>>>(bq, bk, bv, bqkv);
  prep_wo_kernel<<<1024, 256, 0, stream>>>(wo, woT);

  qkv_gemm_kernel<<<dim3(NROW / 16, C3 / 64), 128, 0, stream>>>(
      xh, wt, bqkv, q16, k16, vTd);
  attn_kernel<<<dim3(17, H_, B_), 128, 0, stream>>>(q16, k16, vTd, tz, aout);
  out_gemm_kernel<<<dim3(NROW / 16, F_ / 64), 128, 0, stream>>>(
      aout, woT, bo, out);
}
